// EGNNLayer_41497974014004
// MI455X (gfx1250) — matrix operand order repaired and checked by reading, not compiled, not run on hardware
//
#include <hip/hip_runtime.h>
#include <hip/hip_bf16.h>

// MI455X / gfx1250, wave32. EGNN layer. Edge pass ~13 GFLOP vs ~KBs of input:
// pure compute problem -> all per-edge MLPs via v_wmma_f32_16x16x32_bf16.
// B fragments pre-swizzled into LDS (keeps VGPRs < 256; no s_set_vgpr_msb storm).

typedef __attribute__((ext_vector_type(16))) __bf16 v16bf;
typedef __attribute__((ext_vector_type(8)))  float  v8f;

#define N_PART 1024
#define FDIM   32     // F
#define HDIM   64     // H
#define ODIM   8      // O
#define NFRAG  18

__device__ __forceinline__ float silu_f(float v) {
    return v * __builtin_amdgcn_rcpf(1.0f + __expf(-v));
}

// Wave-scope LDS fence. The "memory" clobber also pins the per-tile LDS
// fragment reloads inside the loop (prevents LICM re-hoisting into VGPRs).
__device__ __forceinline__ void wave_lds_sync() {
    __builtin_amdgcn_wave_barrier();
    asm volatile("s_wait_dscnt 0" ::: "memory");
    __builtin_amdgcn_wave_barrier();
}

__device__ __forceinline__ v8f wmma_bf16(v16bf a, v16bf b, v8f c) {
    return __builtin_amdgcn_wmma_f32_16x16x32_bf16(
        false, a, false, b, (short)0, c, false, false);
}

__device__ __forceinline__ unsigned short bf16_bits(float f) {
    return __builtin_bit_cast(unsigned short, (__bf16)f);
}

// B fragment (32x16 bf16) gathered from a row-major f32 weight matrix.
// Lanes 0-15 hold K={0..7,16..23}, lanes 16-31 hold K={8..15,24..31}.
__device__ __forceinline__ v16bf load_b_frag(const float* W, int ld, int k0, int n0,
                                             int krows, int ncols, int lane) {
    const int col = n0 + (lane & 15);
    const int kb  = k0 + ((lane & 16) ? 8 : 0);
    v16bf b;
#pragma unroll
    for (int idx = 0; idx < 16; ++idx) {
        const int k = kb + idx + ((idx >= 8) ? 8 : 0);
        float v = (k < krows && col < ncols) ? W[k * ld + col] : 0.0f;
        b[idx] = (__bf16)v;
    }
    return b;
}

// A fragment (16x32 bf16) from a column-major bf16 LDS tile buf[col][16 rows].
// Row M = lane&15; lanes<16: K = {0..7,16..23}; lanes>=16: +8.
__device__ __forceinline__ v16bf load_a_cm(const unsigned short* buf, int k0, int lane) {
    const int M   = lane & 15;
    const int kof = (lane & 16) ? 8 : 0;
    v16bf a;
#pragma unroll
    for (int idx = 0; idx < 16; ++idx) {
        const int k = k0 + kof + idx + ((idx >= 8) ? 8 : 0);
        a[idx] = __builtin_bit_cast(__bf16, buf[k * 16 + M]);
    }
    return a;
}

// Pack 8 f32 -> 8 bf16 and store as one b128 into column-major tile.
__device__ __forceinline__ void store_col8(unsigned short* buf, int col, int mbase,
                                           const float* e) {
    uint4 q;
    q.x = (unsigned int)bf16_bits(e[0]) | ((unsigned int)bf16_bits(e[1]) << 16);
    q.y = (unsigned int)bf16_bits(e[2]) | ((unsigned int)bf16_bits(e[3]) << 16);
    q.z = (unsigned int)bf16_bits(e[4]) | ((unsigned int)bf16_bits(e[5]) << 16);
    q.w = (unsigned int)bf16_bits(e[6]) | ((unsigned int)bf16_bits(e[7]) << 16);
    ((uint4*)buf)[col * 2 + (mbase >> 3)] = q;
}

__device__ __forceinline__ float wave_reduce(float v) {
#pragma unroll
    for (int off = 16; off > 0; off >>= 1) v += __shfl_down(v, off, 32);
    return v;
}

// Fragment indices in the LDS fragment cache.
#define F_E1 0   // 4 frags: We1 rows 32..63 (hj), N tiles
#define F_M1 4   // 4 frags: Wm1 (K rows >= 8 zero)
#define F_D1 8   // 4 frags: Wd1
#define F_E2 12  // 2 frags: We2 (cols >= 8 zero), K tiles
#define F_M2 14  // 2 frags: Wm2
#define F_D2 16  // 2 frags: Wd2 (col 0 only)

__global__ __launch_bounds__(256)
void egnn_edge_kernel(const float* __restrict__ x,   const float* __restrict__ h,
                      const float* __restrict__ We1, const float* __restrict__ be1,
                      const float* __restrict__ We2, const float* __restrict__ be2,
                      const float* __restrict__ Wm1, const float* __restrict__ bm1,
                      const float* __restrict__ Wm2, const float* __restrict__ bm2,
                      const float* __restrict__ Wd1, const float* __restrict__ bd1,
                      const float* __restrict__ Wd2, const float* __restrict__ bd2,
                      float* __restrict__ out_x,     float* __restrict__ mpart) {
    __shared__ __align__(32) v16bf          bfrag[NFRAG][32];          // 18 KB
    __shared__ __align__(16) unsigned short e1buf_all[8][HDIM * 16];   // col-major, 16 KB
    __shared__ __align__(16) unsigned short massbuf_all[8][ODIM * 16]; // col-major, 2 KB
    __shared__ __align__(16) float          dsq_all[8][16];
    __shared__ __align__(16) float          dcf_all[8][16];
    __shared__ float hic_lds[HDIM];
    __shared__ float wred_m[8];
    __shared__ float wred_x[8][3];

    const int i_idx = blockIdx.x;
    const int tid   = threadIdx.x;
    const int wave  = tid >> 5;
    const int lane  = tid & 31;
    const int c     = lane & 15;            // output column within 16-wide N tile
    const int mbase = (lane & 16) ? 8 : 0;  // C-layout row base
    const int kofA  = mbase;                // A-layout K offset
    const int M     = lane & 15;

    __builtin_prefetch(We1, 0, 0);
    __builtin_prefetch(Wm1, 0, 0);
    __builtin_prefetch(Wd1, 0, 0);

    // hi contribution: h[i,:] @ We1[0:32,:]  (rank-1 across each j tile)
    if (tid < HDIM) {
        float acc = 0.0f;
        for (int k = 0; k < FDIM; ++k) acc += h[i_idx * FDIM + k] * We1[k * HDIM + tid];
        hic_lds[tid] = acc;
    }

    // Pre-swizzle all 18 B fragments into LDS (each wave builds a subset).
    for (int f = wave; f < NFRAG; f += 8) {
        const float* W; int ld, k0, n0, krows, ncols;
        if (f < 12) {
            const int g = f >> 2, nt = f & 3;
            W  = (g == 0) ? We1 : (g == 1) ? Wm1 : Wd1;
            ld = HDIM; k0 = (g == 0) ? 32 : 0; n0 = nt * 16;
            krows = (g == 0) ? 64 : ODIM; ncols = HDIM;
        } else {
            const int p = f - 12, g = p >> 1, kt = p & 1;
            W  = (g == 0) ? We2 : (g == 1) ? Wm2 : Wd2;
            ld = (g == 2) ? 1 : ODIM; k0 = kt * 32; n0 = 0;
            krows = HDIM; ncols = (g == 2) ? 1 : ODIM;
        }
        bfrag[f][lane] = load_b_frag(W, ld, k0, n0, krows, ncols, lane);
    }
    __syncthreads();

    // Per-lane epilogue constants (column-indexed), kept in VGPRs.
    float hic[4], be1v[4], w64v[4], bm1v[4], bd1v[4];
#pragma unroll
    for (int nt = 0; nt < 4; ++nt) {
        const int col = nt * 16 + c;
        hic[nt]  = hic_lds[col];
        be1v[nt] = be1[col];
        w64v[nt] = We1[64 * HDIM + col];    // dsq row of We1
        bm1v[nt] = bm1[col];
        bd1v[nt] = bd1[col];
    }
    const float be2v = (c < ODIM) ? be2[c] : 0.0f;
    const float bm2v = (c < ODIM) ? bm2[c] : 0.0f;
    const float bd2v = bd2[0];
    const float xi0 = x[i_idx * 3 + 0];
    const float xi1 = x[i_idx * 3 + 1];
    const float xi2 = x[i_idx * 3 + 2];

    unsigned short* e1buf   = e1buf_all[wave];
    unsigned short* massbuf = massbuf_all[wave];
    float*          dsq_s   = dsq_all[wave];
    float*          dcf     = dcf_all[wave];

    float m_loc = 0.0f;
    float xa0 = 0.0f, xa1 = 0.0f, xa2 = 0.0f;

    for (int tt = 0; tt < 8; ++tt) {
        const int jbase = (wave + (tt << 3)) << 4;
        const int j     = jbase + M;
        const float mk  = (j != i_idx) ? 1.0f : 0.0f;

        float d0 = 0.f, d1 = 0.f, d2 = 0.f, inv = 0.f;
        if (lane < 16) {
            d0 = x[j * 3 + 0] - xi0;
            d1 = x[j * 3 + 1] - xi1;
            d2 = x[j * 3 + 2] - xi2;
            const float dsq_raw = d0 * d0 + d1 * d1 + d2 * d2;
            dsq_s[lane] = mk * dsq_raw;
            const float dist = sqrtf(mk > 0.0f ? dsq_raw : 1.0f);
            inv = mk / (dist + 1.0f);
        }
        wave_lds_sync();

        // ---- GEMM1: hj(16x32) @ We1[32:64,:] -> E1 (hi & dsq rank-1) ----
        const float4* hp = (const float4*)(h + j * FDIM);
        const float4 h0 = hp[kofA >> 2];
        const float4 h1 = hp[(kofA >> 2) + 1];
        const float4 h2 = hp[(kofA + 16) >> 2];
        const float4 h3 = hp[((kofA + 16) >> 2) + 1];
        v16bf a1;
        a1[0]  = (__bf16)(mk * h0.x); a1[1]  = (__bf16)(mk * h0.y);
        a1[2]  = (__bf16)(mk * h0.z); a1[3]  = (__bf16)(mk * h0.w);
        a1[4]  = (__bf16)(mk * h1.x); a1[5]  = (__bf16)(mk * h1.y);
        a1[6]  = (__bf16)(mk * h1.z); a1[7]  = (__bf16)(mk * h1.w);
        a1[8]  = (__bf16)(mk * h2.x); a1[9]  = (__bf16)(mk * h2.y);
        a1[10] = (__bf16)(mk * h2.z); a1[11] = (__bf16)(mk * h2.w);
        a1[12] = (__bf16)(mk * h3.x); a1[13] = (__bf16)(mk * h3.y);
        a1[14] = (__bf16)(mk * h3.z); a1[15] = (__bf16)(mk * h3.w);

        const float4 dq0 = ((const float4*)dsq_s)[mbase >> 2];
        const float4 dq1 = ((const float4*)dsq_s)[(mbase >> 2) + 1];
        const float dsqM[8] = {dq0.x, dq0.y, dq0.z, dq0.w, dq1.x, dq1.y, dq1.z, dq1.w};
        float mkr[8];
#pragma unroll
        for (int r = 0; r < 8; ++r)
            mkr[r] = ((jbase + mbase + r) != i_idx) ? 1.0f : 0.0f;

#pragma unroll
        for (int nt = 0; nt < 4; ++nt) {
            v8f acc = {0.f, 0.f, 0.f, 0.f, 0.f, 0.f, 0.f, 0.f};
            acc = wmma_bf16(a1, bfrag[F_E1 + nt][lane], acc);
            float e[8];
#pragma unroll
            for (int r = 0; r < 8; ++r)
                e[r] = silu_f(acc[r] + mkr[r] * hic[nt] + dsqM[r] * w64v[nt] + be1v[nt]);
            store_col8(e1buf, nt * 16 + c, mbase, e);
        }
        wave_lds_sync();

        // ---- GEMM2: E1(16x64) @ We2 -> mass (masked) ----
        v16bf a2a = load_a_cm(e1buf, 0, lane);
        v16bf a2b = load_a_cm(e1buf, 32, lane);
        v8f acc2 = {0.f, 0.f, 0.f, 0.f, 0.f, 0.f, 0.f, 0.f};
        acc2 = wmma_bf16(a2a, bfrag[F_E2 + 0][lane], acc2);
        acc2 = wmma_bf16(a2b, bfrag[F_E2 + 1][lane], acc2);
        float massv[8];
#pragma unroll
        for (int r = 0; r < 8; ++r) massv[r] = mkr[r] * silu_f(acc2[r] + be2v);
        if (c < ODIM) store_col8(massbuf, c, mbase, massv);
        wave_lds_sync();

        // ---- GEMM3: mass(16x8, K zero-padded in regs) @ Wm1 -> M1 ----
        v16bf a3;
#pragma unroll
        for (int idx = 0; idx < 16; ++idx) a3[idx] = (__bf16)0.0f;
        if (lane < 16) {
#pragma unroll
            for (int idx = 0; idx < 8; ++idx)
                a3[idx] = __builtin_bit_cast(__bf16, massbuf[idx * 16 + M]);
        }
#pragma unroll
        for (int nt = 0; nt < 4; ++nt) {
            v8f acc = {0.f, 0.f, 0.f, 0.f, 0.f, 0.f, 0.f, 0.f};
            acc = wmma_bf16(a3, bfrag[F_M1 + nt][lane], acc);
            float e[8];
#pragma unroll
            for (int r = 0; r < 8; ++r) e[r] = silu_f(acc[r] + bm1v[nt]);
            store_col8(e1buf, nt * 16 + c, mbase, e);
        }
        wave_lds_sync();

        // ---- GEMM4: M1 @ Wm2 -> mnet; accumulate global scalar m ----
        v16bf a4a = load_a_cm(e1buf, 0, lane);
        v16bf a4b = load_a_cm(e1buf, 32, lane);
        v8f acc4 = {0.f, 0.f, 0.f, 0.f, 0.f, 0.f, 0.f, 0.f};
        acc4 = wmma_bf16(a4a, bfrag[F_M2 + 0][lane], acc4);
        acc4 = wmma_bf16(a4b, bfrag[F_M2 + 1][lane], acc4);
#pragma unroll
        for (int r = 0; r < 8; ++r) m_loc += massv[r] * silu_f(acc4[r] + bm2v);

        // ---- GEMM5: mass @ Wd1 -> D1 (A fragment reused) ----
#pragma unroll
        for (int nt = 0; nt < 4; ++nt) {
            v8f acc = {0.f, 0.f, 0.f, 0.f, 0.f, 0.f, 0.f, 0.f};
            acc = wmma_bf16(a3, bfrag[F_D1 + nt][lane], acc);
            float e[8];
#pragma unroll
            for (int r = 0; r < 8; ++r) e[r] = silu_f(acc[r] + bd1v[nt]);
            store_col8(e1buf, nt * 16 + c, mbase, e);
        }
        wave_lds_sync();

        // ---- GEMM6: D1 @ Wd2 -> dcoef (column 0) ----
        v16bf a6a = load_a_cm(e1buf, 0, lane);
        v16bf a6b = load_a_cm(e1buf, 32, lane);
        v8f acc6 = {0.f, 0.f, 0.f, 0.f, 0.f, 0.f, 0.f, 0.f};
        acc6 = wmma_bf16(a6a, bfrag[F_D2 + 0][lane], acc6);
        acc6 = wmma_bf16(a6b, bfrag[F_D2 + 1][lane], acc6);
        if (c == 0) {
            float4 qa, qb;
            qa.x = silu_f(acc6[0] + bd2v); qa.y = silu_f(acc6[1] + bd2v);
            qa.z = silu_f(acc6[2] + bd2v); qa.w = silu_f(acc6[3] + bd2v);
            qb.x = silu_f(acc6[4] + bd2v); qb.y = silu_f(acc6[5] + bd2v);
            qb.z = silu_f(acc6[6] + bd2v); qb.w = silu_f(acc6[7] + bd2v);
            ((float4*)dcf)[mbase >> 2]       = qa;
            ((float4*)dcf)[(mbase >> 2) + 1] = qb;
        }
        wave_lds_sync();

        // ---- coordinate-update accumulation (lanes 0..15 own one edge) ----
        if (lane < 16) {
            const float s = dcf[lane] * inv;   // inv already carries the mask
            xa0 += d0 * s; xa1 += d1 * s; xa2 += d2 * s;
        }
        wave_lds_sync();
    }

    // Deterministic reductions: wave shuffle tree -> LDS -> serial sum.
    const float msum = wave_reduce(m_loc);
    const float x0s = wave_reduce(xa0);
    const float x1s = wave_reduce(xa1);
    const float x2s = wave_reduce(xa2);
    if (lane == 0) {
        wred_m[wave]    = msum;
        wred_x[wave][0] = x0s;
        wred_x[wave][1] = x1s;
        wred_x[wave][2] = x2s;
    }
    __syncthreads();
    if (tid == 0) {
        float s = 0.0f;
        for (int w = 0; w < 8; ++w) s += wred_m[w];
        mpart[i_idx] = s;
    }
    if (tid < 3) {
        float s = 0.0f;
        for (int w = 0; w < 8; ++w) s += wred_x[w][tid];
        out_x[i_idx * 3 + tid] = x[i_idx * 3 + tid] + s;
    }
}

__global__ __launch_bounds__(256)
void egnn_node_kernel(const float* __restrict__ h, const float* __restrict__ Wh1,
                      const float* __restrict__ bh1, const float* __restrict__ mpart,
                      float* __restrict__ out_h) {
    __shared__ float red[256];
    const int tid = threadIdx.x;
    float s = 0.0f;
    for (int k = tid; k < N_PART; k += 256) s += mpart[k];
    red[tid] = s;
    __syncthreads();
    for (int off = 128; off > 0; off >>= 1) {
        if (tid < off) red[tid] += red[tid + off];
        __syncthreads();
    }
    const float m = red[0];

    const int gid = blockIdx.x * 256 + tid;   // N_PART * HDIM threads total
    const int i = gid >> 6, n = gid & 63;
    float acc = bh1[n];
    for (int k = 0; k < FDIM; ++k) acc += h[i * FDIM + k] * Wh1[k * HDIM + n];
    acc += m * Wh1[FDIM * HDIM + n];
    out_h[gid] = silu_f(acc);
}

extern "C" void kernel_launch(void* const* d_in, const int* in_sizes, int n_in,
                              void* d_out, int out_size, void* d_ws, size_t ws_size,
                              hipStream_t stream) {
    (void)in_sizes; (void)n_in; (void)out_size; (void)ws_size;
    const float* x   = (const float*)d_in[0];
    const float* h   = (const float*)d_in[1];
    const float* We1 = (const float*)d_in[2];
    const float* be1 = (const float*)d_in[3];
    const float* We2 = (const float*)d_in[4];
    const float* be2 = (const float*)d_in[5];
    const float* Wm1 = (const float*)d_in[6];
    const float* bm1 = (const float*)d_in[7];
    const float* Wm2 = (const float*)d_in[8];
    const float* bm2 = (const float*)d_in[9];
    const float* Wd1 = (const float*)d_in[10];
    const float* bd1 = (const float*)d_in[11];
    const float* Wd2 = (const float*)d_in[12];
    const float* bd2 = (const float*)d_in[13];
    const float* Wh1 = (const float*)d_in[14];
    const float* bh1 = (const float*)d_in[15];

    float* out_x = (float*)d_out;               // [1024,3]
    float* out_h = out_x + N_PART * 3;          // [1024,64]
    float* mpart = (float*)d_ws;                // [1024] per-block partial of m

    egnn_edge_kernel<<<dim3(N_PART), dim3(256), 0, stream>>>(
        x, h, We1, be1, We2, be2, Wm1, bm1, Wm2, bm2,
        Wd1, bd1, Wd2, bd2, out_x, mpart);
    egnn_node_kernel<<<dim3((N_PART * HDIM) / 256), dim3(256), 0, stream>>>(
        h, Wh1, bh1, mpart, out_h);
}
